// QuanvolutionClassifier_65481071402942
// MI455X (gfx1250) — matrix operand order repaired and checked by reading, not compiled, not run on hardware
//
#include <hip/hip_runtime.h>
#include <hip/hip_bf16.h>

// ---------------------------------------------------------------------------
// QuanvolutionClassifier for MI455X (gfx1250)
//   inputs : x[4096,784] f32, theta[2,4] f32, W1[784,128] f32, b1[128] f32,
//            W2[128,10] f32, b2[10] f32
//   output : log_softmax(relu(quanv(x) @ W1 + b1) @ W2 + b2)  -> [4096,10] f32
//
// Pipeline (all intermediates are L2-resident, <8.1 MB total):
//   1) quanv_kernel : 1 thread per (image, patch); 16-amp state sim in regs,
//      writes feats as f16 [4096 x 800] (K padded 784 -> 800 for WMMA).
//   2) pack_w1 / pack_w2 : transpose+pad weights to N-major f16 so the WMMA
//      B fragment is two 16B loads per lane (k+0..7, k+16..23).
//   3) gemm1 : v_wmma_f32_16x16x32_f16, 16x16 tile per wave, K=800 (25 steps),
//      epilogue bias+ReLU -> h f16 [4096 x 128].
//   4) gemm2 : same, K=128 (4 steps), logits f32 [4096 x 16] (10 valid cols).
//   5) lsm_kernel : per-row log-softmax over the 10 valid columns.
// ---------------------------------------------------------------------------

typedef __attribute__((ext_vector_type(16))) _Float16 v16h;
typedef __attribute__((ext_vector_type(8)))  _Float16 v8h;
typedef __attribute__((ext_vector_type(8)))  float    v8f;

#define BATCH   4096
#define KPAD    800     // 784 padded to multiple of 32
#define HID     128
#define NOUT    10
#define NPAD    16

// ---------------- circuit helpers (state idx bit: wire w -> bit 3-w) --------

__device__ __forceinline__ void ry_gate(float s[16], int bit, float c, float sn) {
    const int m = 1 << bit;
#pragma unroll
    for (int i = 0; i < 16; ++i) {
        if (!(i & m)) {
            float a = s[i], b = s[i | m];
            s[i]     = c * a - sn * b;
            s[i | m] = sn * a + c * b;
        }
    }
}

__device__ __forceinline__ void cnot_gate(float s[16], int cbit, int tbit) {
    const int cm = 1 << cbit, tm = 1 << tbit;
#pragma unroll
    for (int i = 0; i < 16; ++i) {
        if ((i & cm) && !(i & tm)) {
            float t = s[i];
            s[i]       = s[i | tm];
            s[i | tm]  = t;
        }
    }
}

// ---------------- 1) quanvolution: one thread per (b, patch) ----------------
// p in [0,196): real patch. p in [196,200): writes the 16 zero pad columns.

__global__ void quanv_kernel(const float* __restrict__ x,
                             const float* __restrict__ theta,
                             _Float16* __restrict__ feats) {
    const int PP = 200;
    int gid = blockIdx.x * blockDim.x + threadIdx.x;
    if (gid >= BATCH * PP) return;
    int b = gid / PP;
    int p = gid - b * PP;
    _Float16* frow = feats + (size_t)b * KPAD;

    if (p >= 196) {                       // zero the K-pad columns 784..799
        int c0 = 784 + 4 * (p - 196);
        frow[c0 + 0] = (_Float16)0.f; frow[c0 + 1] = (_Float16)0.f;
        frow[c0 + 2] = (_Float16)0.f; frow[c0 + 3] = (_Float16)0.f;
        return;
    }

    int r = p / 14, c = p - r * 14;
    const float* xr = x + (size_t)b * 784;
    float ang[4];
    ang[0] = xr[(2 * r) * 28 + 2 * c];
    ang[1] = xr[(2 * r) * 28 + 2 * c + 1];
    ang[2] = xr[(2 * r + 1) * 28 + 2 * c];
    ang[3] = xr[(2 * r + 1) * 28 + 2 * c + 1];

    float s[16];
#pragma unroll
    for (int i = 0; i < 16; ++i) s[i] = 0.f;
    s[0] = 1.f;

    // encoder: RY(x_w) on wire w
#pragma unroll
    for (int w = 0; w < 4; ++w) {
        float sn, cs;
        __sincosf(0.5f * ang[w], &sn, &cs);
        ry_gate(s, 3 - w, cs, sn);
    }
    // 2 variational layers: RY(theta) per wire then CNOT ring
#pragma unroll
    for (int l = 0; l < 2; ++l) {
#pragma unroll
        for (int w = 0; w < 4; ++w) {
            float sn, cs;
            __sincosf(0.5f * theta[l * 4 + w], &sn, &cs);
            ry_gate(s, 3 - w, cs, sn);
        }
        cnot_gate(s, 3, 2);   // (0,1)
        cnot_gate(s, 2, 1);   // (1,2)
        cnot_gate(s, 1, 0);   // (2,3)
        cnot_gate(s, 0, 3);   // (3,0)
    }
    // <Z_w> = sum_i (+/-) |s_i|^2
#pragma unroll
    for (int w = 0; w < 4; ++w) {
        const int m = 1 << (3 - w);
        float z = 0.f;
#pragma unroll
        for (int i = 0; i < 16; ++i) {
            float p2 = s[i] * s[i];
            z += (i & m) ? -p2 : p2;
        }
        frow[4 * p + w] = (_Float16)z;
    }
}

// ---------------- 2) weight pack/transpose to f16, N-major ------------------

__global__ void pack_w1_kernel(const float* __restrict__ W1,  // [784,128]
                               _Float16* __restrict__ w1t) {  // [128,800]
    int idx = blockIdx.x * blockDim.x + threadIdx.x;
    if (idx >= HID * KPAD) return;
    int n = idx / KPAD;
    int k = idx - n * KPAD;
    w1t[idx] = (k < 784) ? (_Float16)W1[(size_t)k * HID + n] : (_Float16)0.f;
}

__global__ void pack_w2_kernel(const float* __restrict__ W2,  // [128,10]
                               _Float16* __restrict__ w2t) {  // [16,128]
    int idx = blockIdx.x * blockDim.x + threadIdx.x;
    if (idx >= NPAD * HID) return;
    int n = idx / HID;
    int k = idx - n * HID;
    w2t[idx] = (n < NOUT) ? (_Float16)W2[(size_t)k * NOUT + n] : (_Float16)0.f;
}

// ---------------- fragment loaders (ISA 16-bit A/B layouts) -----------------
// A 16x32: lane-half h holds K = {k0+8h .. k0+8h+7} U {k0+8h+16 .. k0+8h+23}
// B 32x16: lane-half h holds K = {k0+8h .. k0+8h+7} U {k0+8h+16 .. k0+8h+23}

__device__ __forceinline__ v16h ld_frag_a(const _Float16* p) {
    v8h lo = *(const v8h*)(p);
    v8h hi = *(const v8h*)(p + 16);
    return __builtin_shufflevector(lo, hi, 0, 1, 2, 3, 4, 5, 6, 7,
                                           8, 9, 10, 11, 12, 13, 14, 15);
}

__device__ __forceinline__ v16h ld_frag_b(const _Float16* p) {
    v8h lo = *(const v8h*)(p);
    v8h hi = *(const v8h*)(p + 16);
    return __builtin_shufflevector(lo, hi, 0, 1, 2, 3, 4, 5, 6, 7,
                                           8, 9, 10, 11, 12, 13, 14, 15);
}

// ---------------- 3) GEMM1: feats[4096x800] @ W1 -> relu -> h f16 -----------
// 4 waves / block, one 16x16 tile per wave. 256 M-tiles x 8 N-tiles.

__global__ void gemm1_kernel(const _Float16* __restrict__ feats,
                             const _Float16* __restrict__ w1t,
                             const float* __restrict__ b1,
                             _Float16* __restrict__ h) {
    int wave  = threadIdx.x >> 5;
    int lane  = threadIdx.x & 31;
    int tile  = blockIdx.x * 4 + wave;
    int tm    = tile >> 3;            // 0..255
    int tn    = tile & 7;             // 0..7
    int lhalf = lane >> 4;            // 0/1
    int lidx  = lane & 15;

    const _Float16* arow = feats + (size_t)(tm * 16 + lidx) * KPAD + (lhalf << 3);
    const _Float16* brow = w1t   + (size_t)(tn * 16 + lidx) * KPAD + (lhalf << 3);

    v8f acc = {};
    for (int k0 = 0; k0 < KPAD; k0 += 32) {
        v16h a = ld_frag_a(arow + k0);
        v16h b = ld_frag_b(brow + k0);
        acc = __builtin_amdgcn_wmma_f32_16x16x32_f16(
            false, a, false, b, (short)0, acc, false, false);
    }

    float bias = b1[tn * 16 + lidx];
    int row0 = tm * 16 + (lhalf << 3);   // C layout: VGPR i -> M = i + 8*lhalf
    int col  = tn * 16 + lidx;           //           N = lane & 15
#pragma unroll
    for (int i = 0; i < 8; ++i) {
        float v = acc[i] + bias;
        v = v > 0.f ? v : 0.f;
        h[(size_t)(row0 + i) * HID + col] = (_Float16)v;
    }
}

// ---------------- 4) GEMM2: h[4096x128] @ W2 -> logits f32 [4096x16] --------

__global__ void gemm2_kernel(const _Float16* __restrict__ h,
                             const _Float16* __restrict__ w2t,
                             float* __restrict__ logits) {
    int wave  = threadIdx.x >> 5;
    int lane  = threadIdx.x & 31;
    int tm    = blockIdx.x * 4 + wave;  // 0..255 (single N tile)
    int lhalf = lane >> 4;
    int lidx  = lane & 15;

    const _Float16* arow = h   + (size_t)(tm * 16 + lidx) * HID + (lhalf << 3);
    const _Float16* brow = w2t + (size_t)lidx * HID + (lhalf << 3);

    v8f acc = {};
#pragma unroll
    for (int k0 = 0; k0 < HID; k0 += 32) {
        v16h a = ld_frag_a(arow + k0);
        v16h b = ld_frag_b(brow + k0);
        acc = __builtin_amdgcn_wmma_f32_16x16x32_f16(
            false, a, false, b, (short)0, acc, false, false);
    }

    int row0 = tm * 16 + (lhalf << 3);
#pragma unroll
    for (int i = 0; i < 8; ++i)
        logits[(size_t)(row0 + i) * NPAD + lidx] = acc[i];
}

// ---------------- 5) log-softmax over the 10 valid columns ------------------

__global__ void lsm_kernel(const float* __restrict__ logits,
                           const float* __restrict__ b2,
                           float* __restrict__ out) {
    int r = blockIdx.x * blockDim.x + threadIdx.x;
    if (r >= BATCH) return;
    float l[NOUT];
    float m = -3.0e38f;
#pragma unroll
    for (int j = 0; j < NOUT; ++j) {
        l[j] = logits[(size_t)r * NPAD + j] + b2[j];
        m = l[j] > m ? l[j] : m;
    }
    float sum = 0.f;
#pragma unroll
    for (int j = 0; j < NOUT; ++j) sum += __expf(l[j] - m);
    float lse = __logf(sum);
#pragma unroll
    for (int j = 0; j < NOUT; ++j) out[(size_t)r * NOUT + j] = l[j] - m - lse;
}

// ---------------------------------------------------------------------------

extern "C" void kernel_launch(void* const* d_in, const int* in_sizes, int n_in,
                              void* d_out, int out_size, void* d_ws, size_t ws_size,
                              hipStream_t stream) {
    const float* x     = (const float*)d_in[0];
    const float* theta = (const float*)d_in[1];
    const float* W1    = (const float*)d_in[2];
    const float* b1    = (const float*)d_in[3];
    const float* W2    = (const float*)d_in[4];
    const float* b2    = (const float*)d_in[5];
    float* out         = (float*)d_out;

    // workspace layout (all offsets 256B aligned)
    char* ws = (char*)d_ws;
    _Float16* feats = (_Float16*)(ws);                       // 4096*800*2  = 6,553,600
    _Float16* w1t   = (_Float16*)(ws + 6553600);             // 128*800*2   =   204,800
    _Float16* w2t   = (_Float16*)(ws + 6758400);             // 16*128*2    =     4,096
    _Float16* h     = (_Float16*)(ws + 6762496);             // 4096*128*2  = 1,048,576
    float*    lg    = (float*)   (ws + 7811072);             // 4096*16*4   =   262,144

    // 1) quanvolution features (+K padding)
    {
        int n = BATCH * 200;
        quanv_kernel<<<(n + 255) / 256, 256, 0, stream>>>(x, theta, feats);
    }
    // 2) weight packing
    {
        int n1 = HID * KPAD;   // 102400
        pack_w1_kernel<<<(n1 + 255) / 256, 256, 0, stream>>>(W1, w1t);
        int n2 = NPAD * HID;   // 2048
        pack_w2_kernel<<<(n2 + 255) / 256, 256, 0, stream>>>(W2, w2t);
    }
    // 3) GEMM1 + bias + ReLU : 2048 tiles, 4 waves/block
    gemm1_kernel<<<512, 128, 0, stream>>>(feats, w1t, b1, h);
    // 4) GEMM2 : 256 tiles, 4 waves/block
    gemm2_kernel<<<64, 128, 0, stream>>>(h, w2t, lg);
    // 5) log-softmax
    lsm_kernel<<<(BATCH + 255) / 256, 256, 0, stream>>>(lg, b2, out);
}
